// LSTMModel_91027536871576
// MI455X (gfx1250) — matrix operand order repaired and checked by reading, not compiled, not run on hardware
//
#include <hip/hip_runtime.h>

// ---------------------------------------------------------------------------
// LSTM LM for MI455X (gfx1250, wave32, WMMA 16x16x32 bf16 -> f32 acc)
//
// T=256, B=16, D=512, V=50257.
// Phase 0: gather X = E[idx] (bf16), convert Wx/Uh/Wout to bf16.
// Phase 1: Xg = X @ Wx^T + bx  (parallel WMMA GEMM, [4096,512]x[512,2048], f32 out)
// Phase 2: sequential recurrence: gates = Xg[t] + h@Uh^T + bh, cell math,
//          h kept in LDS, c in registers; one 1024-thread workgroup, 32 waves,
//          wave w owns d-chunk w*16..w*16+15 across all 4 gates.
//          Prefetch (global_prefetch_b8) hides Xg[t+1] latency behind WMMAs.
// Phase 3: logits = H @ Wout^T + b (big WMMA GEMM, [4096,512]x[512,50257]),
//          Wout bf16 (51 MB) is L2-resident on the 192 MB L2.
// Phase 4: per-row online logsumexp; Phase 5: subtract -> log_softmax.
// ---------------------------------------------------------------------------

#define TT 256
#define BB 16
#define DD 512
#define VV 50257
#define MROWS (TT * BB)        // 4096
#define GCOLS (4 * DD)         // 2048
#define NT_V 3142              // ceil(V/16)

typedef __attribute__((ext_vector_type(16))) __bf16 bf16x16;
typedef __attribute__((ext_vector_type(8)))  float  floatx8;

union FragB16 { bf16x16 v; uint4 q[2]; };

static __device__ __forceinline__ __bf16 f2bf(float x) {
  unsigned u = __builtin_bit_cast(unsigned, x);
  unsigned r = u + 0x7FFFu + ((u >> 16) & 1u);   // round-to-nearest-even
  unsigned short s = (unsigned short)(r >> 16);
  return __builtin_bit_cast(__bf16, s);
}

// A fragment (16xK row-major, 16-bit): lane m = L&15, hi = L>>4.
// halves j=0..15 -> K = k0+hi*16+j. Two 16B loads.
static __device__ __forceinline__ bf16x16 load_a_frag(const __bf16* base, int ld,
                                                      int row0, int k0, int lane) {
  int m = lane & 15, hi = lane >> 4;
  const __bf16* p = base + (size_t)(row0 + m) * ld + k0 + hi * 16;
  FragB16 f;
  f.q[0] = *(const uint4*)p;
  f.q[1] = *(const uint4*)(p + 8);
  return f.v;
}

// B fragment (Kx16, source stored row-major [N][K] i.e. weight[out][in]):
// lane n = L&15 = column, hi = L>>4; halves j -> K = k0 + hi*16 + j. Two 16B loads.
static __device__ __forceinline__ bf16x16 load_b_frag(const __bf16* base, int ld,
                                                      int col0, int k0, int lane) {
  int n = lane & 15, hi = lane >> 4;
  const __bf16* p = base + (size_t)(col0 + n) * ld + k0 + hi * 16;
  FragB16 f;
  f.q[0] = *(const uint4*)p;
  f.q[1] = *(const uint4*)(p + 8);
  return f.v;
}

// ---------------- Phase 0 kernels ----------------
__global__ void k_cvt_bf16(const float* __restrict__ src, __bf16* __restrict__ dst,
                           long n) {
  long i = (long)blockIdx.x * blockDim.x + threadIdx.x;
  long stride = (long)gridDim.x * blockDim.x;
  for (; i < n; i += stride) dst[i] = f2bf(src[i]);
}

__global__ void k_gather_x(const int* __restrict__ idx, const float* __restrict__ E,
                           __bf16* __restrict__ X) {
  long i = (long)blockIdx.x * blockDim.x + threadIdx.x;   // over 4096*512
  int m = (int)(i >> 9);
  int d = (int)(i & 511);
  X[i] = f2bf(E[(size_t)idx[m] * DD + d]);
}

// ---------------- Phase 1: Xg = X @ Wx^T + bx ----------------
__global__ __launch_bounds__(256) void k_gemm_xg(const __bf16* __restrict__ X,
                                                 const __bf16* __restrict__ Wxb,
                                                 const float* __restrict__ bx,
                                                 float* __restrict__ Xg) {
  int tid = threadIdx.x, lane = tid & 31, wv = tid >> 5;
  int w = blockIdx.x * 8 + wv;            // 32768 waves
  int nt = w & 127, mt = w >> 7;
  int row0 = mt * 16, col0 = nt * 16;
  int n = lane & 15, hi = lane >> 4;
  float bias = bx[col0 + n];
  floatx8 acc;
#pragma unroll
  for (int r = 0; r < 8; ++r) acc[r] = bias;
  for (int k0 = 0; k0 < DD; k0 += 32) {
    bf16x16 a = load_a_frag(X, DD, row0, k0, lane);
    bf16x16 b = load_b_frag(Wxb, DD, col0, k0, lane);  // flat [2048][512]
    acc = __builtin_amdgcn_wmma_f32_16x16x32_bf16(false, a, false, b,
                                                  (short)0, acc, false, false);
  }
#pragma unroll
  for (int r = 0; r < 8; ++r)
    Xg[(size_t)(row0 + hi * 8 + r) * GCOLS + col0 + n] = acc[r];
}

// ---------------- Phase 2: sequential recurrence ----------------
__global__ __launch_bounds__(1024) void k_lstm(const float* __restrict__ Xg,
                                               const __bf16* __restrict__ Uhb,
                                               const float* __restrict__ bh,
                                               __bf16* __restrict__ H) {
  __shared__ __bf16 hbuf[BB * DD];        // 16 KB: h_t, rows = batch
  int tid = threadIdx.x, lane = tid & 31, wv = tid >> 5;
  int n = lane & 15, hi = lane >> 4;
  int dcol = wv * 16 + n;                 // this lane's d index

  for (int i = tid; i < BB * DD; i += 1024) hbuf[i] = f2bf(0.f);

  float creg[8];
#pragma unroll
  for (int r = 0; r < 8; ++r) creg[r] = 0.f;
  float bgate[4];
#pragma unroll
  for (int g = 0; g < 4; ++g) bgate[g] = bh[g * DD + dcol];
  __syncthreads();

  for (int t = 0; t < TT; ++t) {
    int row0 = t * BB;
    floatx8 acc[4];
#pragma unroll
    for (int g = 0; g < 4; ++g) {
      int col0 = g * DD + wv * 16;
#pragma unroll
      for (int r = 0; r < 8; ++r)
        acc[g][r] = Xg[(size_t)(row0 + hi * 8 + r) * GCOLS + col0 + n];
    }
    // Prefetch next step's Xg tiles (lowers to global_prefetch_b8) so the
    // dependent-step accumulator fetch is hidden behind this step's WMMAs.
    if (t + 1 < TT) {
#pragma unroll
      for (int g = 0; g < 4; ++g) {
        int col0 = g * DD + wv * 16;
#pragma unroll
        for (int r = 0; r < 8; ++r)
          __builtin_prefetch(&Xg[(size_t)(row0 + BB + hi * 8 + r) * GCOLS + col0 + n],
                             0, 3);
      }
    }
    for (int k0 = 0; k0 < DD; k0 += 32) {
      FragB16 fa;
      fa.q[0] = *(const uint4*)&hbuf[n * DD + k0 + hi * 16];
      fa.q[1] = *(const uint4*)&hbuf[n * DD + k0 + hi * 16 + 8];
#pragma unroll
      for (int g = 0; g < 4; ++g) {
        int col0 = g * DD + wv * 16;
        FragB16 fb;
        const __bf16* pb = Uhb + (size_t)(col0 + n) * DD + k0 + hi * 16;
        fb.q[0] = *(const uint4*)pb;
        fb.q[1] = *(const uint4*)(pb + 8);
        acc[g] = __builtin_amdgcn_wmma_f32_16x16x32_bf16(false, fa.v, false, fb.v,
                                                         (short)0, acc[g],
                                                         false, false);
      }
    }
    __syncthreads();    // all reads of hbuf done before overwrite
#pragma unroll
    for (int r = 0; r < 8; ++r) {
      int b = hi * 8 + r;
      float fg = 1.f / (1.f + __expf(-(acc[0][r] + bgate[0])));
      float ig = 1.f / (1.f + __expf(-(acc[1][r] + bgate[1])));
      float og = 1.f / (1.f + __expf(-(acc[2][r] + bgate[2])));
      float gg = tanhf(acc[3][r] + bgate[3]);
      creg[r] = fg * creg[r] + ig * gg;
      float hn = og * tanhf(creg[r]);
      __bf16 hb = f2bf(hn);
      hbuf[b * DD + dcol] = hb;
      H[(size_t)(row0 + b) * DD + dcol] = hb;
    }
    __syncthreads();    // new h visible before next step's A loads
  }
}

// ---------------- Phase 3: logits = H @ Wout^T + b ----------------
// Each wave: 4 M-tiles (64 rows) x 1 N-tile; B-frag reused 4x per K step.
__global__ __launch_bounds__(256) void k_logits(const __bf16* __restrict__ H,
                                                const __bf16* __restrict__ Wob,
                                                const float* __restrict__ wbias,
                                                float* __restrict__ out) {
  int tid = threadIdx.x, lane = tid & 31, wv = tid >> 5;
  int w = blockIdx.x * 8 + wv;            // 64 * 3142 waves, exact grid
  int mg = w / NT_V;
  int nt = w - mg * NT_V;
  int row0 = mg * 64, col0 = nt * 16;
  int n = lane & 15, hi = lane >> 4;
  int col = col0 + n;
  bool cv = col < VV;
  float bias = cv ? wbias[col] : 0.f;

  // Warm WGP$ for this wave's Wout column stripe (speculative, L2-resident).
  if (cv) __builtin_prefetch(Wob + (size_t)col * DD, 0, 3);

  floatx8 acc[4];
#pragma unroll
  for (int j = 0; j < 4; ++j)
#pragma unroll
    for (int r = 0; r < 8; ++r) acc[j][r] = bias;

  for (int k0 = 0; k0 < DD; k0 += 32) {
    FragB16 fb;
    if (cv) {
      const __bf16* p = Wob + (size_t)col * DD + k0 + hi * 16;
      fb.q[0] = *(const uint4*)p;
      fb.q[1] = *(const uint4*)(p + 8);
    } else {
      fb.q[0] = make_uint4(0, 0, 0, 0);
      fb.q[1] = make_uint4(0, 0, 0, 0);
    }
#pragma unroll
    for (int j = 0; j < 4; ++j) {
      bf16x16 a = load_a_frag(H, DD, row0 + j * 16, k0, lane);
      acc[j] = __builtin_amdgcn_wmma_f32_16x16x32_bf16(false, a, false, fb.v,
                                                       (short)0, acc[j],
                                                       false, false);
    }
  }
  if (cv) {
#pragma unroll
    for (int j = 0; j < 4; ++j)
#pragma unroll
      for (int r = 0; r < 8; ++r)
        out[(size_t)(row0 + j * 16 + hi * 8 + r) * VV + col] = acc[j][r];
  }
}

// ---------------- Phase 4: per-row logsumexp (online) ----------------
__global__ __launch_bounds__(256) void k_lse(const float* __restrict__ out,
                                             float* __restrict__ lse) {
  int row = blockIdx.x, tid = threadIdx.x;
  const float* p = out + (size_t)row * VV;
  float m = -__builtin_inff(), s = 0.f;
  for (int v = tid; v < VV; v += 256) {
    float x = p[v];
    if (x > m) { s = s * __expf(m - x) + 1.f; m = x; }
    else       { s += __expf(x - m); }
  }
  __shared__ float sm[256], ss[256];
  sm[tid] = m; ss[tid] = s;
  __syncthreads();
  for (int off = 128; off; off >>= 1) {
    if (tid < off) {
      float m2 = sm[tid + off], s2 = ss[tid + off];
      float M = fmaxf(sm[tid], m2);
      ss[tid] = ss[tid] * __expf(sm[tid] - M) + s2 * __expf(m2 - M);
      sm[tid] = M;
    }
    __syncthreads();
  }
  if (tid == 0) lse[row] = sm[0] + __logf(ss[0]);
}

// ---------------- Phase 5: subtract lse ----------------
__global__ __launch_bounds__(256) void k_sub(float* __restrict__ out,
                                             const float* __restrict__ lse) {
  int v = blockIdx.x * 256 + threadIdx.x;
  int row = blockIdx.y;
  if (v < VV) out[(size_t)row * VV + v] -= lse[row];
}

extern "C" void kernel_launch(void* const* d_in, const int* in_sizes, int n_in,
                              void* d_out, int out_size, void* d_ws, size_t ws_size,
                              hipStream_t stream) {
  const int*   idx    = (const int*)d_in[0];
  const float* E      = (const float*)d_in[1];
  const float* Wx     = (const float*)d_in[2];
  const float* bx     = (const float*)d_in[3];
  const float* Uh     = (const float*)d_in[4];
  const float* bh     = (const float*)d_in[5];
  const float* Wout_w = (const float*)d_in[6];
  const float* Wout_b = (const float*)d_in[7];
  float* out = (float*)d_out;
  char* ws = (char*)d_ws;

  // workspace layout (256B aligned)
  __bf16* X    = (__bf16*)(ws + 0);                      //  4,194,304
  __bf16* Wxb  = (__bf16*)(ws + 4194304);                //  2,097,152
  __bf16* Uhb  = (__bf16*)(ws + 6291456);                //  2,097,152
  __bf16* Wob  = (__bf16*)(ws + 8388608);                // 51,463,168
  float*  Xg   = (float*)(ws + 59851776);                // 33,554,432
  __bf16* H    = (__bf16*)(ws + 93406208);               //  4,194,304
  float*  lse  = (float*)(ws + 97600512);                //     16,384

  // Phase 0: conversions + embedding gather
  k_cvt_bf16<<<4096, 256, 0, stream>>>(Wx, Wxb, (long)4 * DD * DD);
  k_cvt_bf16<<<4096, 256, 0, stream>>>(Uh, Uhb, (long)4 * DD * DD);
  k_cvt_bf16<<<100514, 256, 0, stream>>>(Wout_w, Wob, (long)VV * DD);
  k_gather_x<<<8192, 256, 0, stream>>>(idx, E, X);

  // Phase 1: input-gate GEMM (recurrence-independent)
  k_gemm_xg<<<4096, 256, 0, stream>>>(X, Wxb, bx, Xg);

  // Phase 2: sequential LSTM on a single WGP-resident workgroup
  k_lstm<<<1, 1024, 0, stream>>>(Xg, Uhb, bh, H);

  // Phase 3: output projection, whole-chip WMMA GEMM (Wout L2-resident)
  k_logits<<<25136, 256, 0, stream>>>(H, Wob, Wout_b, out);

  // Phase 4+5: log_softmax
  k_lse<<<MROWS, 256, 0, stream>>>(out, lse);
  dim3 gsub((VV + 255) / 256, MROWS);
  k_sub<<<gsub, 256, 0, stream>>>(out, lse);
}
